// Diffuser_283467842401
// MI455X (gfx1250) — matrix operand order repaired and checked by reading, not compiled, not run on hardware
//
#include <hip/hip_runtime.h>
#include <hip/hip_bf16.h>
#include <cstdint>
#include <cstddef>

#define A_  512
#define F_  80
#define P_  20
#define M_  20000
#define S_  4
#define D_  128
#define E_  400000
#define NF_ (A_*F_)
#define NP_ (A_*P_)

typedef __bf16 bf16;
typedef __attribute__((ext_vector_type(16))) __bf16 v16bf;
typedef __attribute__((ext_vector_type(8)))  float  v8f;

// ---------------------------------------------------------------- utilities
__global__ void k_fill0(int* p, int n) {
    int i = blockIdx.x * 256 + threadIdx.x;
    if (i < n) p[i] = 0;
}
__global__ void k_copy_i(const int* a, int* b, int n) {
    int i = blockIdx.x * 256 + threadIdx.x;
    if (i < n) b[i] = a[i];
}
__global__ void k_hist(const int* dst, int* cnt) {
    int i = blockIdx.x * 256 + threadIdx.x;
    if (i < E_) atomicAdd(&cnt[dst[i]], 1);
}
__global__ void k_scan_block(const int* in, int* out, int* bsum, int n) {
    __shared__ int sm[256];
    int t = threadIdx.x;
    int i = blockIdx.x * 256 + t;
    int v = (i < n) ? in[i] : 0;
    sm[t] = v;
    __syncthreads();
    for (int off = 1; off < 256; off <<= 1) {
        int add = (t >= off) ? sm[t - off] : 0;
        __syncthreads();
        sm[t] += add;
        __syncthreads();
    }
    if (i < n) out[i] = sm[t] - v;          // exclusive
    if (t == 255) bsum[blockIdx.x] = sm[255];
}
__global__ void k_scan_carry(const int* bsum, int* boff, int nb) {
    if (threadIdx.x == 0 && blockIdx.x == 0) {
        int run = 0;
        for (int i = 0; i < nb; i++) { int v = bsum[i]; boff[i] = run; run += v; }
    }
}
__global__ void k_scan_add(int* out, const int* boff, int n) {
    int i = blockIdx.x * 256 + threadIdx.x;
    if (i < n) out[i] += boff[blockIdx.x];
}
__global__ void k_scatter(const int* dst, int* cur, int* perm) {
    int i = blockIdx.x * 256 + threadIdx.x;
    if (i < E_) { int p = atomicAdd(&cur[dst[i]], 1); perm[p] = i; }
}
__global__ void k_permute(const int* perm, const int* src, const float* feat, int nc,
                          int* srcS, float* featS) {
    int i = blockIdx.x * 256 + threadIdx.x;
    if (i < E_) {
        int e = perm[i];
        srcS[i] = src[e];
        for (int c = 0; c < nc; c++) featS[(size_t)i * nc + c] = feat[(size_t)e * nc + c];
    }
}
// W [K=128][N=128] f32 -> Wt [N][K] bf16
__global__ void k_transpose(const float* W, bf16* Wt) {
    int n = blockIdx.x, k = threadIdx.x;
    Wt[n * 128 + k] = (bf16)W[k * 128 + n];
}

// ---------------------------------------------------------------- LayerNorm
__global__ __launch_bounds__(128) void k_ln(const float* __restrict__ x,
                                            const float* __restrict__ g,
                                            const float* __restrict__ b,
                                            bf16* __restrict__ out) {
    int r = blockIdx.x, t = threadIdx.x;
    float v = x[(size_t)r * 128 + t];
    float s = v;
    for (int o = 16; o >= 1; o >>= 1) s += __shfl_xor(s, o, 32);
    __shared__ float red[4];
    if ((t & 31) == 0) red[t >> 5] = s;
    __syncthreads();
    float mean = (red[0] + red[1] + red[2] + red[3]) * (1.f / 128.f);
    float d = v - mean;
    float s2 = d * d;
    for (int o = 16; o >= 1; o >>= 1) s2 += __shfl_xor(s2, o, 32);
    __syncthreads();
    if ((t & 31) == 0) red[t >> 5] = s2;
    __syncthreads();
    float var = (red[0] + red[1] + red[2] + red[3]) * (1.f / 128.f);
    out[(size_t)r * 128 + t] = (bf16)(d * rsqrtf(var + 1e-5f) * g[t] + b[t]);
}

// ----------------------------------------------- time embedding (+ te2 = temb @ fw1[4:] + fb1)
__global__ __launch_bounds__(128) void k_time(const float* __restrict__ dt,
                                              const float* __restrict__ tw1, const float* __restrict__ tb1,
                                              const float* __restrict__ tw2, const float* __restrict__ tb2,
                                              const float* __restrict__ fw1, const float* __restrict__ fb1,
                                              float* __restrict__ te2) {
    __shared__ float emb[128], h1[128], tr[128];
    int a = blockIdx.x, t = threadIdx.x;
    float tv = dt[a];
    const float c = -9.210340371976184f / 63.f;   // -ln(10000)/(half-1)
    if (t < 64) emb[t] = sinf(tv * __expf((float)t * c));
    else        emb[t] = cosf(tv * __expf((float)(t - 64) * c));
    __syncthreads();
    float acc = tb1[t];
    for (int i = 0; i < 128; i++) acc += emb[i] * tw1[i * 128 + t];
    h1[t] = acc / (1.f + __expf(-acc));           // silu
    __syncthreads();
    acc = tb2[t];
    for (int i = 0; i < 128; i++) acc += h1[i] * tw2[i * 128 + t];
    tr[t] = acc;
    __syncthreads();
    acc = fb1[t];
    for (int i = 0; i < 128; i++) acc += tr[i] * fw1[(4 + i) * 128 + t];
    te2[a * 128 + t] = acc;
}

// future pre: h = te2[a] + x_t[n,:4]@fw1[:4]; LN; silu -> bf16
__global__ __launch_bounds__(128) void k_future_pre(const float* __restrict__ x_t,
                                                    const float* __restrict__ te2,
                                                    const float* __restrict__ fw1,
                                                    const float* __restrict__ lng,
                                                    const float* __restrict__ lnb,
                                                    bf16* __restrict__ out) {
    int n = blockIdx.x, t = threadIdx.x;
    int a = n / F_;
    float h = te2[a * 128 + t];
    const float* xr = x_t + (size_t)n * 4;
    for (int i = 0; i < 4; i++) h += xr[i] * fw1[i * 128 + t];
    float s = h;
    for (int o = 16; o >= 1; o >>= 1) s += __shfl_xor(s, o, 32);
    __shared__ float red[4];
    if ((t & 31) == 0) red[t >> 5] = s;
    __syncthreads();
    float mean = (red[0] + red[1] + red[2] + red[3]) * (1.f / 128.f);
    float d = h - mean;
    float s2 = d * d;
    for (int o = 16; o >= 1; o >>= 1) s2 += __shfl_xor(s2, o, 32);
    __syncthreads();
    if ((t & 31) == 0) red[t >> 5] = s2;
    __syncthreads();
    float var = (red[0] + red[1] + red[2] + red[3]) * (1.f / 128.f);
    float y = d * rsqrtf(var + 1e-5f) * lng[t] + lnb[t];
    out[(size_t)n * 128 + t] = (bf16)(y / (1.f + __expf(-y)));
}

// edge-attr first layer: relu(feat @ w1 + b1) -> bf16
__global__ void k_edge_mlp1(const float* __restrict__ featS, int nc,
                            const float* __restrict__ w1, const float* __restrict__ b1,
                            bf16* __restrict__ out) {
    size_t idx = (size_t)blockIdx.x * 256 + threadIdx.x;
    size_t e = idx >> 7;
    int j = (int)(idx & 127);
    if (e < (size_t)E_) {
        float acc = b1[j];
        for (int c = 0; c < nc; c++) acc += featS[e * nc + c] * w1[c * 128 + j];
        out[idx] = (bf16)fmaxf(acc, 0.f);
    }
}

// ---------------------------------------------------------------- WMMA GEMM
// C[M,128] = act(A[M,128](bf16) @ Wt^T + bias) (+resid), Wt is [N=128][K=128] bf16.
// Weights staged in LDS (136-elem padded rows -> conflict-free 32B fragments).
// 256 threads = 8 waves, 128 rows/block. B-fragments software-pipelined (double buffer)
// so each ds_load waits are overlapped with the previous v_wmma.
#define LWS 136
template <int ACT, int RES, int OUTF32, int BIAS>
__global__ __launch_bounds__(256) void k_gemm128(const bf16* __restrict__ A,
                                                 const bf16* __restrict__ Wt,
                                                 const float* __restrict__ bias,
                                                 const float* __restrict__ resid,
                                                 void* __restrict__ outp, int Mrows) {
    __shared__ __align__(16) bf16 lw[128 * LWS];
    const int tid = threadIdx.x;
    {   // cooperative stage of 128x128 bf16 weights into padded LDS
        int n = tid >> 1, h = (tid & 1) * 64;
        const uint4* s = (const uint4*)(Wt + n * 128 + h);
        uint4* d = (uint4*)(lw + n * LWS + h);
#pragma unroll
        for (int i = 0; i < 8; i++) d[i] = s[i];
    }
    __syncthreads();

    const int w = tid >> 5, lane = tid & 31;
    const int lh = lane >> 4, ll = lane & 15;
    const int mtile = blockIdx.x * 128 + w * 16;
    int r = mtile + ll;
    bool oob = (r >= Mrows);
    const bf16* arow = A + (size_t)(oob ? 0 : r) * 128;
    const bf16* lbase = lw + ll * LWS + lh * 8;

    uint4 z4; z4.x = z4.y = z4.z = z4.w = 0u;

    // preload all four A fragments (one clause of global_load_b128)
    v16bf af[4];
#pragma unroll
    for (int ks = 0; ks < 4; ++ks) {
        uint4 alo = *(const uint4*)(arow + ks * 32 + lh * 8);
        uint4 ahi = *(const uint4*)(arow + ks * 32 + 16 + lh * 8);
        if (oob) { alo = z4; ahi = z4; }
        ((uint4*)&af[ks])[0] = alo;
        ((uint4*)&af[ks])[1] = ahi;
    }

    v8f acc[8];
#pragma unroll
    for (int n = 0; n < 8; n++)
#pragma unroll
        for (int i = 0; i < 8; i++) acc[n][i] = 0.f;

    // B fragment for (ks,n): lbase + n*16*LWS + ks*32
    auto loadB = [&](int t) {
        const bf16* bp = lbase + (t & 7) * (16 * LWS) + (t >> 3) * 32;
        v16bf b;
        ((uint4*)&b)[0] = *(const uint4*)bp;
        ((uint4*)&b)[1] = *(const uint4*)(bp + 16);
        return b;
    };

    v16bf bc = loadB(0);
#pragma unroll
    for (int t = 0; t < 32; ++t) {
        v16bf bn = loadB(t < 31 ? t + 1 : t);     // prefetch next tile's B
        acc[t & 7] = __builtin_amdgcn_wmma_f32_16x16x32_bf16(
            false, af[t >> 3], false, bc, (short)0, acc[t & 7], false, false);
        bc = bn;
    }

    const int rbase = mtile + lh * 8;
#pragma unroll
    for (int n = 0; n < 8; n++) {
        const int col = n * 16 + ll;
        float bi = BIAS ? bias[col] : 0.f;
#pragma unroll
        for (int v = 0; v < 8; v++) {
            int row = rbase + v;
            if (row < Mrows) {
                float x = acc[n][v] + bi;
                if (RES) x += resid[(size_t)row * 128 + col];
                if (ACT == 1) x = fmaxf(x, 0.f);
                if (ACT == 2) x = x / (1.f + __expf(-x));
                if (OUTF32) ((float*)outp)[(size_t)row * 128 + col] = x;
                else        ((bf16*)outp)[(size_t)row * 128 + col] = (bf16)x;
            }
        }
    }
}

// ------------------------------------------- per-dst online segment-softmax attention
// block = one dst node (128 threads = 8 heads x 16 dims), CSR edge list
__global__ __launch_bounds__(128) void k_attn(const int* __restrict__ rowptr,
                                              const int* __restrict__ srcS,
                                              const bf16* __restrict__ Qn,
                                              const bf16* __restrict__ Kn,
                                              const bf16* __restrict__ Vn,
                                              const bf16* __restrict__ eak,
                                              const bf16* __restrict__ eav,
                                              bf16* __restrict__ aggb) {
    int d = blockIdx.x, t = threadIdx.x;
    float q = (float)Qn[(size_t)d * 128 + t] * 0.25f;   // 1/sqrt(DH=16)
    int e0 = rowptr[d], e1 = rowptr[d + 1];
    float m = -3.4e38f, z = 0.f, acc = 0.f;
    for (int e = e0; e < e1; e++) {
        int s = srcS[e];
        size_t eo = (size_t)e * 128 + t;
        size_t so = (size_t)s * 128 + t;
        float k = (float)Kn[so] + (float)eak[eo];
        float p = q * k;
        p += __shfl_xor(p, 8, 16);
        p += __shfl_xor(p, 4, 16);
        p += __shfl_xor(p, 2, 16);
        p += __shfl_xor(p, 1, 16);
        float v = (float)Vn[so] + (float)eav[eo];
        float mn = fmaxf(m, p);
        float sc = __expf(m - mn);
        float pe = __expf(p - mn);
        z = z * sc + pe;
        acc = acc * sc + pe * v;
        m = mn;
    }
    aggb[(size_t)d * 128 + t] = (bf16)(acc / (z + 1e-9f));
}

// final projection: out[n,s] = h2[n]@w2[:,s] + b2[s]   (w2 [128,4])
__global__ void k_outproj(const bf16* __restrict__ h2, const float* __restrict__ w2,
                          const float* __restrict__ b2, float* __restrict__ out) {
    int idx = blockIdx.x * 256 + threadIdx.x;
    if (idx < NF_ * S_) {
        int s = idx & 3;
        size_t n = (size_t)(idx >> 2);
        float acc = b2[s];
        const bf16* hr = h2 + n * 128;
        for (int i = 0; i < 128; i++) acc += (float)hr[i] * w2[i * 4 + s];
        out[idx] = acc;
    }
}

// ================================================================ host
extern "C" void kernel_launch(void* const* d_in, const int* in_sizes, int n_in,
                              void* d_out, int out_size, void* d_ws, size_t ws_size,
                              hipStream_t stream) {
    (void)in_sizes; (void)n_in; (void)out_size; (void)ws_size;
    const float* x_t     = (const float*)d_in[0];
    const float* dtime   = (const float*)d_in[1];
    const float* past    = (const float*)d_in[2];
    const float* mapE    = (const float*)d_in[3];
    const float* feat_pf = (const float*)d_in[4];
    const float* feat_mf = (const float*)d_in[5];
    const float* feat_ff = (const float*)d_in[6];
    const int* src_pf = (const int*)d_in[7];
    const int* dst_pf = (const int*)d_in[8];
    const int* src_mf = (const int*)d_in[9];
    const int* dst_mf = (const int*)d_in[10];
    const int* src_ff = (const int*)d_in[11];
    const int* dst_ff = (const int*)d_in[12];
    auto PF = [&](int i) { return (const float*)d_in[i]; };
    // params flattened (jax key-sorted): attn(3 layers x {ff,mf,pf} x 10), future(6),
    // mlp_ff(4), mlp_mf(4), mlp_pf(4), out(6), time(4)
    // attn fields: Wk=0,Wke=1,Wo=2,Wq=3,Wv=4,Wve=5,ln_dst_b=6,ln_dst_g=7,ln_src_b=8,ln_src_g=9

    uint8_t* wp = (uint8_t*)d_ws;
    auto alloc = [&](size_t bytes) -> void* {
        void* r = wp;
        wp += (bytes + 255) & ~(size_t)255;
        return r;
    };
    bf16* ea[3];
    for (int r = 0; r < 3; r++) ea[r] = (bf16*)alloc((size_t)E_ * 128 * 2);
    bf16* eak = (bf16*)alloc((size_t)E_ * 128 * 2);
    bf16* eav = (bf16*)alloc((size_t)E_ * 128 * 2);
    bf16* htmp = eak;                       // alias: used before eak is needed
    float* feat_s = (float*)alloc((size_t)E_ * 6 * 4);
    int* srcS[3];
    for (int r = 0; r < 3; r++) srcS[r] = (int*)alloc((size_t)E_ * 4);
    int* rowptr[3];
    for (int r = 0; r < 3; r++) rowptr[r] = (int*)alloc((size_t)(NF_ + 1) * 4);
    int* perm = (int*)alloc((size_t)E_ * 4);
    int* cnt  = (int*)alloc((size_t)(NF_ + 1) * 4);
    int* bsum = (int*)alloc(1024);
    int* boff = (int*)alloc(1024);
    float* te2    = (float*)alloc((size_t)A_ * 128 * 4);
    float* future = (float*)alloc((size_t)NF_ * 128 * 4);
    bf16* h2   = (bf16*)alloc((size_t)NF_ * 128 * 2);
    bf16* hs   = (bf16*)alloc((size_t)NF_ * 128 * 2);
    bf16* hd   = (bf16*)alloc((size_t)NF_ * 128 * 2);
    bf16* Qn   = (bf16*)alloc((size_t)NF_ * 128 * 2);
    bf16* Kn   = (bf16*)alloc((size_t)NF_ * 128 * 2);
    bf16* Vn   = (bf16*)alloc((size_t)NF_ * 128 * 2);
    bf16* aggb = (bf16*)alloc((size_t)NF_ * 128 * 2);
    bf16* Wt[7];
    for (int i = 0; i < 7; i++) Wt[i] = (bf16*)alloc(128 * 128 * 2);

    const int nb = (NF_ + 1 + 255) / 256;      // scan chunks
    const int eb = (E_ + 255) / 256;

    // ---- time embedding + future embed
    k_time<<<A_, 128, 0, stream>>>(dtime, PF(129), PF(127), PF(130), PF(128),
                                   PF(107), PF(103), te2);
    k_future_pre<<<NF_, 128, 0, stream>>>(x_t, te2, PF(107), PF(106), PF(105), h2);
    k_transpose<<<128, 128, 0, stream>>>(PF(108), Wt[0]);
    k_gemm128<0, 0, 1, 1><<<NF_ / 128, 256, 0, stream>>>(h2, Wt[0], PF(104), nullptr,
                                                         future, NF_);

    // ---- per-relation edge prep: CSR sort + permute + edge-attr MLP
    auto prep_rel = [&](int r, const int* dst, const int* src, const float* feat, int nc,
                        const float* w1, const float* b1, const float* w2, const float* b2) {
        k_fill0<<<nb, 256, 0, stream>>>(cnt, NF_ + 1);
        k_hist<<<eb, 256, 0, stream>>>(dst, cnt);
        k_scan_block<<<nb, 256, 0, stream>>>(cnt, rowptr[r], bsum, NF_ + 1);
        k_scan_carry<<<1, 32, 0, stream>>>(bsum, boff, nb);
        k_scan_add<<<nb, 256, 0, stream>>>(rowptr[r], boff, NF_ + 1);
        k_copy_i<<<nb, 256, 0, stream>>>(rowptr[r], cnt, NF_ + 1);
        k_scatter<<<eb, 256, 0, stream>>>(dst, cnt, perm);
        k_permute<<<eb, 256, 0, stream>>>(perm, src, feat, nc, srcS[r], feat_s);
        k_edge_mlp1<<<(int)(((size_t)E_ * 128) / 256), 256, 0, stream>>>(feat_s, nc, w1, b1, htmp);
        k_transpose<<<128, 128, 0, stream>>>(w2, Wt[0]);
        k_gemm128<0, 0, 0, 1><<<E_ / 128, 256, 0, stream>>>(htmp, Wt[0], b2, nullptr, ea[r], E_);
    };
    prep_rel(0, dst_pf, src_pf, feat_pf, 6, PF(119), PF(117), PF(120), PF(118)); // mlp_pf
    prep_rel(1, dst_mf, src_mf, feat_mf, 6, PF(115), PF(113), PF(116), PF(114)); // mlp_mf
    prep_rel(2, dst_ff, src_ff, feat_ff, 5, PF(111), PF(109), PF(112), PF(110)); // mlp_ff

    // ---- attention blocks
    auto attn_block = [&](int layer, int relA, const float* x_src, int n_src,
                          const int* sS, const int* rp, const bf16* eaR) {
        auto ai = [&](int f) { return 13 + (layer * 3 + relA) * 10 + f; };
        k_transpose<<<128, 128, 0, stream>>>(PF(ai(3)), Wt[1]);  // Wq
        k_transpose<<<128, 128, 0, stream>>>(PF(ai(0)), Wt[2]);  // Wk
        k_transpose<<<128, 128, 0, stream>>>(PF(ai(4)), Wt[3]);  // Wv
        k_transpose<<<128, 128, 0, stream>>>(PF(ai(1)), Wt[4]);  // Wke
        k_transpose<<<128, 128, 0, stream>>>(PF(ai(5)), Wt[5]);  // Wve
        k_transpose<<<128, 128, 0, stream>>>(PF(ai(2)), Wt[6]);  // Wo
        k_ln<<<n_src, 128, 0, stream>>>(x_src, PF(ai(9)), PF(ai(8)), hs);
        k_ln<<<NF_, 128, 0, stream>>>(future, PF(ai(7)), PF(ai(6)), hd);
        const int gs = (n_src + 127) / 128;
        k_gemm128<0, 0, 0, 0><<<NF_ / 128, 256, 0, stream>>>(hd, Wt[1], nullptr, nullptr, Qn, NF_);
        k_gemm128<0, 0, 0, 0><<<gs, 256, 0, stream>>>(hs, Wt[2], nullptr, nullptr, Kn, n_src);
        k_gemm128<0, 0, 0, 0><<<gs, 256, 0, stream>>>(hs, Wt[3], nullptr, nullptr, Vn, n_src);
        k_gemm128<0, 0, 0, 0><<<E_ / 128, 256, 0, stream>>>(eaR, Wt[4], nullptr, nullptr, eak, E_);
        k_gemm128<0, 0, 0, 0><<<E_ / 128, 256, 0, stream>>>(eaR, Wt[5], nullptr, nullptr, eav, E_);
        k_attn<<<NF_, 128, 0, stream>>>(rp, sS, Qn, Kn, Vn, eak, eav, aggb);
        k_gemm128<0, 1, 1, 0><<<NF_ / 128, 256, 0, stream>>>(aggb, Wt[6], nullptr, future,
                                                             future, NF_);
    };
    for (int l = 0; l < 3; l++) {
        attn_block(l, 2, past,   NP_, srcS[0], rowptr[0], ea[0]);   // pf
        attn_block(l, 1, mapE,   M_,  srcS[1], rowptr[1], ea[1]);   // mf
        attn_block(l, 0, future, NF_, srcS[2], rowptr[2], ea[2]);   // ff
    }

    // ---- output head: LN -> silu(h@w1+b1) -> @w2+b2
    k_ln<<<NF_, 128, 0, stream>>>(future, PF(124), PF(123), hd);
    k_transpose<<<128, 128, 0, stream>>>(PF(125), Wt[0]);
    k_gemm128<2, 0, 0, 1><<<NF_ / 128, 256, 0, stream>>>(hd, Wt[0], PF(121), nullptr, h2, NF_);
    k_outproj<<<(NF_ * S_ + 255) / 256, 256, 0, stream>>>(h2, PF(126), PF(122), (float*)d_out);
}
